// MAE3D_67104569033403
// MI455X (gfx1250) — compile-verified, matrix operand order checked by reading, not run on hardware
//
#include <hip/hip_runtime.h>
#include <hip/hip_bf16.h>

#define B_ 8
#define N_ 2048
#define KN_ 20
#define P_ 64
#define PS_ 32
#define D_ 1024
#define MLP_ 2048
#define DEPTH_ 6
#define H_ 8
#define DH_ 64
#define INNER_ 512
#define L_ 65
#define LP_ 528          // 520 tokens padded to a multiple of 16
#define EPS_ 1e-5f

typedef __hip_bfloat16 bf16;
typedef __attribute__((ext_vector_type(16))) __bf16 v16bf;
typedef __attribute__((ext_vector_type(8)))  float v8f;

__device__ __forceinline__ float leaky_(float v){ return v >= 0.f ? v : 0.2f*v; }

// ---------------------------------------------------------------------------
// Guard-free bf16 WMMA GEMM:  C[M,N] = A[M,K] * BT[N,K]^T (+Cin) (+bias) (act)
// Requirements (enforced by caller via padding):
//   M % 16 == 0, N % 64 == 0, K % 32 == 0, lda/ldb % 8 == 0 (16B alignment).
// One wave computes a 16x64 tile: 1 A fragment x 4 B fragments -> 4 WMMAs/K-step.
// A-fragment dwords u[0..3] = k[half*8 .. half*8+7]  -> one b128 load
//            dwords u[4..7] = k[16+half*8 .. +7]     -> one b128 load
// ---------------------------------------------------------------------------
__global__ __launch_bounds__(256)
void gemm_k(const bf16* __restrict__ A, int lda, long sA,
            const bf16* __restrict__ BT, int ldb, long sB,
            float* __restrict__ C, int ldc, long sC,
            const float* __restrict__ Cin,
            const float* __restrict__ bias,
            bf16* __restrict__ Dbf,
            int M, int Nn, int K, int act)
{
    const int lane = threadIdx.x & 31;
    const int wid  = threadIdx.x >> 5;
    const int tm   = blockIdx.x * 8 + wid;
    if (tm >= (M >> 4)) return;
    const int tn0  = blockIdx.y << 2;      // first of 4 n-tiles (64 columns)
    const long zb  = blockIdx.z;
    const int half = lane >> 4;
    const int lcol = lane & 15;

    const bf16* arow = A  + zb*sA + (size_t)(tm*16 + lcol)*lda + (half << 3);
    const bf16* br0  = BT + zb*sB + (size_t)((tn0+0)*16 + lcol)*ldb + (half << 3);
    const bf16* br1  = BT + zb*sB + (size_t)((tn0+1)*16 + lcol)*ldb + (half << 3);
    const bf16* br2  = BT + zb*sB + (size_t)((tn0+2)*16 + lcol)*ldb + (half << 3);
    const bf16* br3  = BT + zb*sB + (size_t)((tn0+3)*16 + lcol)*ldb + (half << 3);

    union Acc { v8f v; float f[8]; } c0, c1, c2, c3;
    if (Cin) {
        const float* Ci = Cin + zb*sC;
        #pragma unroll
        for (int v = 0; v < 8; ++v) {
            const size_t m = (size_t)(tm*16 + half*8 + v);
            c0.f[v] = Ci[m*ldc + (tn0+0)*16 + lcol];
            c1.f[v] = Ci[m*ldc + (tn0+1)*16 + lcol];
            c2.f[v] = Ci[m*ldc + (tn0+2)*16 + lcol];
            c3.f[v] = Ci[m*ldc + (tn0+3)*16 + lcol];
        }
    } else {
        #pragma unroll
        for (int v = 0; v < 8; ++v) { c0.f[v]=0.f; c1.f[v]=0.f; c2.f[v]=0.f; c3.f[v]=0.f; }
    }

    union Frag { v16bf v; uint4 q[2]; };
    for (int kk = 0; kk < K; kk += 32) {
        Frag a, f0, f1, f2, f3;
        a.q[0]  = *(const uint4*)(arow + kk);
        a.q[1]  = *(const uint4*)(arow + kk + 16);
        f0.q[0] = *(const uint4*)(br0 + kk);  f0.q[1] = *(const uint4*)(br0 + kk + 16);
        f1.q[0] = *(const uint4*)(br1 + kk);  f1.q[1] = *(const uint4*)(br1 + kk + 16);
        f2.q[0] = *(const uint4*)(br2 + kk);  f2.q[1] = *(const uint4*)(br2 + kk + 16);
        f3.q[0] = *(const uint4*)(br3 + kk);  f3.q[1] = *(const uint4*)(br3 + kk + 16);
        __builtin_prefetch((const void*)(arow + kk + 64), 0, 0); // speculative, next K-slab
        c0.v = __builtin_amdgcn_wmma_f32_16x16x32_bf16(false, a.v, false, f0.v, (short)0, c0.v, false, false);
        c1.v = __builtin_amdgcn_wmma_f32_16x16x32_bf16(false, a.v, false, f1.v, (short)0, c1.v, false, false);
        c2.v = __builtin_amdgcn_wmma_f32_16x16x32_bf16(false, a.v, false, f2.v, (short)0, c2.v, false, false);
        c3.v = __builtin_amdgcn_wmma_f32_16x16x32_bf16(false, a.v, false, f3.v, (short)0, c3.v, false, false);
    }

    float* Cb = C + zb*sC;
    #pragma unroll
    for (int j = 0; j < 4; ++j) {
        const Acc* cj = (j==0)?&c0:(j==1)?&c1:(j==2)?&c2:&c3;
        const int n = (tn0 + j)*16 + lcol;
        const float bvv = bias ? bias[n] : 0.f;
        #pragma unroll
        for (int v = 0; v < 8; ++v) {
            const size_t m = (size_t)(tm*16 + half*8 + v);
            float val = cj->f[v] + bvv;
            if (act == 2) val = 0.5f * val * (1.f + erff(val * 0.70710678118f)); // exact GELU
            Cb[m*ldc + n] = val;
            if (Dbf) Dbf[m*ldc + n] = __float2bfloat16(val);
        }
    }
}

// ------------------------- small helper kernels ----------------------------
__global__ void cast_k(const float* __restrict__ s, bf16* __restrict__ d, size_t n) {
    for (size_t i = (size_t)blockIdx.x*blockDim.x + threadIdx.x; i < n; i += (size_t)gridDim.x*blockDim.x)
        d[i] = __float2bfloat16(s[i]);
}
// cast rows x cols into a zero-padded [rows, ldd] bf16 buffer
__global__ void castpad_k(const float* __restrict__ s, bf16* __restrict__ d,
                          int rows, int cols, int ldd) {
    size_t tot = (size_t)rows * ldd;
    for (size_t i = (size_t)blockIdx.x*blockDim.x + threadIdx.x; i < tot; i += (size_t)gridDim.x*blockDim.x) {
        int c = (int)(i % ldd), r = (int)(i / ldd);
        d[i] = __float2bfloat16(c < cols ? s[(size_t)r*cols + c] : 0.f);
    }
}
__global__ void trcast_k(const float* __restrict__ W, bf16* __restrict__ WT, int Kd, int Nd) {
    size_t tot = (size_t)Kd * Nd;
    for (size_t i = (size_t)blockIdx.x*blockDim.x + threadIdx.x; i < tot; i += (size_t)gridDim.x*blockDim.x) {
        int nn = (int)(i % Nd), kk = (int)(i / Nd);
        WT[(size_t)nn*Kd + kk] = __float2bfloat16(W[i]);
    }
}
__global__ void zero_k(float* __restrict__ p, size_t n) {
    for (size_t i = (size_t)blockIdx.x*blockDim.x + threadIdx.x; i < n; i += (size_t)gridDim.x*blockDim.x)
        p[i] = 0.f;
}
__global__ void diag_k(const float* __restrict__ gram, float* __restrict__ diag) {
    size_t tot = (size_t)B_*N_;
    for (size_t i = (size_t)blockIdx.x*blockDim.x + threadIdx.x; i < tot; i += (size_t)gridDim.x*blockDim.x) {
        size_t b = i / N_, n = i % N_;
        diag[i] = gram[b*(size_t)N_*N_ + n*(size_t)N_ + n];
    }
}
// top-20 of dist(m) = 2*G[n,m] - G[n,n] - G[m,m]; one thread per point row
__global__ void topk_k(const float* __restrict__ gram, const float* __restrict__ diag,
                       int* __restrict__ idx) {
    size_t tot = (size_t)B_*N_;
    for (size_t i = (size_t)blockIdx.x*blockDim.x + threadIdx.x; i < tot; i += (size_t)gridDim.x*blockDim.x) {
        size_t b = i / N_, n = i % N_;
        const float* g  = gram + b*(size_t)N_*N_ + n*(size_t)N_;
        const float* dg = diag + b*(size_t)N_;
        const float dn = dg[n];
        float vals[KN_]; int ids[KN_];
        #pragma unroll
        for (int t = 0; t < KN_; ++t) { vals[t] = -3.0e38f; ids[t] = 0; }
        for (int m = 0; m < N_; ++m) {
            float d = 2.f*g[m] - dn - dg[m];
            if (d > vals[KN_-1]) {
                int j = KN_-1;
                while (j > 0 && vals[j-1] < d) { vals[j] = vals[j-1]; ids[j] = ids[j-1]; --j; }
                vals[j] = d; ids[j] = m;
            }
        }
        #pragma unroll
        for (int t = 0; t < KN_; ++t) idx[i*KN_ + t] = ids[t];
    }
}
// graph feature row (ld-padded): F[.., :C]=nbr-ctr, F[.., C:2C]=ctr
__global__ void graph_k(const float* __restrict__ xf, const int* __restrict__ idx,
                        bf16* __restrict__ F, int C, int ld) {
    size_t tot = (size_t)B_*N_*KN_*C;
    for (size_t i = (size_t)blockIdx.x*blockDim.x + threadIdx.x; i < tot; i += (size_t)gridDim.x*blockDim.x) {
        int c = (int)(i % C); size_t r = i / C;
        size_t bn = r / KN_;
        size_t b = bn / N_;
        int kk = (int)(r % KN_);
        int nb = idx[bn*KN_ + kk];
        float ctr = xf[bn*(size_t)C + c];
        float nbv = xf[(b*(size_t)N_ + nb)*(size_t)C + c];
        bf16* o = F + r*(size_t)ld;
        o[c]     = __float2bfloat16(nbv - ctr);
        o[C + c] = __float2bfloat16(ctr);
    }
}
// BN stats: per-channel sum/sumsq over M rows of y[M,O] -> sums[0:O], sums[O:2O]
__global__ void bnpart_k(const float* __restrict__ y, long M, int O, float* __restrict__ sums) {
    const int o = blockIdx.x;
    long rows = (M + gridDim.y - 1) / gridDim.y;
    long r0 = (long)blockIdx.y * rows;
    long r1 = r0 + rows; if (r1 > M) r1 = M;
    float s = 0.f, s2 = 0.f;
    for (long r = r0 + threadIdx.x; r < r1; r += blockDim.x) {
        float v = y[(size_t)r*O + o]; s += v; s2 += v*v;
    }
    __shared__ float ra[256], rb[256];
    ra[threadIdx.x] = s; rb[threadIdx.x] = s2; __syncthreads();
    for (int st = 128; st > 0; st >>= 1) {
        if (threadIdx.x < st) { ra[threadIdx.x] += ra[threadIdx.x+st]; rb[threadIdx.x] += rb[threadIdx.x+st]; }
        __syncthreads();
    }
    if (threadIdx.x == 0) { atomicAdd(&sums[o], ra[0]); atomicAdd(&sums[O+o], rb[0]); }
}
__global__ void bnfin_k(const float* __restrict__ sums, const float* __restrict__ g,
                        const float* __restrict__ b, float* __restrict__ scale,
                        float* __restrict__ shift, int O, float cnt) {
    int o = blockIdx.x*blockDim.x + threadIdx.x;
    if (o < O) {
        float mu  = sums[o] / cnt;
        float var = sums[O+o] / cnt - mu*mu;
        float sc  = g[o] * rsqrtf(var + EPS_);
        scale[o] = sc; shift[o] = b[o] - mu*sc;
    }
}
// BN-apply + leaky + max over k ; writes f32 [B,N,O] and bf16 slice of concat h
__global__ void efuse_k(const float* __restrict__ Y, const float* __restrict__ scale,
                        const float* __restrict__ shift, float* __restrict__ xf,
                        bf16* __restrict__ hsl, int O) {
    size_t tot = (size_t)B_*N_*O;
    for (size_t i = (size_t)blockIdx.x*blockDim.x + threadIdx.x; i < tot; i += (size_t)gridDim.x*blockDim.x) {
        int o = (int)(i % O); size_t bn = i / O;
        float sc = scale[o], sh = shift[o];
        const float* yr = Y + bn*(size_t)(KN_*O) + o;
        float best = -3.0e38f;
        #pragma unroll 4
        for (int kk = 0; kk < KN_; ++kk) best = fmaxf(best, leaky_(yr[(size_t)kk*O]*sc + sh));
        if (xf) xf[bn*(size_t)O + o] = best;
        hsl[bn*512 + o] = __float2bfloat16(best);
    }
}
// BN-apply + leaky + patch max-pool into t[:,1:,:]
__global__ void pool_k(const float* __restrict__ y5, const float* __restrict__ scale,
                       const float* __restrict__ shift, float* __restrict__ t) {
    size_t tot = (size_t)B_*P_*D_;
    for (size_t i = (size_t)blockIdx.x*blockDim.x + threadIdx.x; i < tot; i += (size_t)gridDim.x*blockDim.x) {
        int d = (int)(i % D_); size_t r = i / D_;
        int p = (int)(r % P_); size_t b = r / P_;
        float sc = scale[d], sh = shift[d];
        float best = -3.0e38f;
        for (int s = 0; s < PS_; ++s) {
            size_t row = b*(size_t)N_ + (size_t)p*PS_ + s;
            best = fmaxf(best, leaky_(y5[row*(size_t)D_ + d]*sc + sh));
        }
        t[(b*(size_t)L_ + p + 1)*(size_t)D_ + d] = best;
    }
}
__global__ void cls_k(const float* __restrict__ cls, float* __restrict__ t) {
    size_t tot = (size_t)B_*D_;
    for (size_t i = (size_t)blockIdx.x*blockDim.x + threadIdx.x; i < tot; i += (size_t)gridDim.x*blockDim.x)
        t[(i / D_)*(size_t)L_*D_ + (i % D_)] = cls[i % D_];
}
__global__ void addpos_k(float* __restrict__ t, const float* __restrict__ pos) {
    size_t tot = (size_t)B_*L_*D_;
    for (size_t i = (size_t)blockIdx.x*blockDim.x + threadIdx.x; i < tot; i += (size_t)gridDim.x*blockDim.x)
        t[i] += pos[i];
}
__global__ __launch_bounds__(256)
void ln_k(const float* __restrict__ x, const float* __restrict__ g,
          const float* __restrict__ b, bf16* __restrict__ y) {
    const int row = blockIdx.x;
    const float* xr = x + (size_t)row * D_;
    float s = 0.f, s2 = 0.f;
    for (int c = threadIdx.x; c < D_; c += 256) { float v = xr[c]; s += v; s2 += v*v; }
    __shared__ float ra[256], rb[256];
    ra[threadIdx.x] = s; rb[threadIdx.x] = s2; __syncthreads();
    for (int st = 128; st > 0; st >>= 1) {
        if (threadIdx.x < st) { ra[threadIdx.x] += ra[threadIdx.x+st]; rb[threadIdx.x] += rb[threadIdx.x+st]; }
        __syncthreads();
    }
    float mu  = ra[0] * (1.f/D_);
    float var = rb[0] * (1.f/D_) - mu*mu;
    float inv = rsqrtf(var + EPS_);
    for (int c = threadIdx.x; c < D_; c += 256)
        y[(size_t)row*D_ + c] = __float2bfloat16((xr[c]-mu)*inv*g[c] + b[c]);
}
// Flash-style attention over 65 tokens; one block per (b,h)
__global__ __launch_bounds__(128)
void attn_k(const float* __restrict__ qkv, bf16* __restrict__ z) {
    const int b = blockIdx.x / H_, h = blockIdx.x % H_;
    __shared__ float kb[L_*DH_], vb[L_*DH_];
    for (int i = threadIdx.x; i < L_*DH_; i += 128) {
        int l = i >> 6, d = i & 63;
        size_t base = ((size_t)(b*L_ + l)*3);
        kb[i] = qkv[((base + 1)*H_ + h)*DH_ + d];
        vb[i] = qkv[((base + 2)*H_ + h)*DH_ + d];
    }
    __syncthreads();
    const int i = threadIdx.x;
    if (i < L_) {
        float q[DH_], acc[DH_];
        #pragma unroll
        for (int d = 0; d < DH_; ++d) {
            q[d] = qkv[(((size_t)(b*L_ + i)*3 + 0)*H_ + h)*DH_ + d] * 0.125f;
            acc[d] = 0.f;
        }
        float m = -3.0e38f, lsum = 0.f;
        for (int j = 0; j < L_; ++j) {
            float s = 0.f;
            #pragma unroll
            for (int d = 0; d < DH_; ++d) s += q[d] * kb[j*DH_ + d];
            float mn = fmaxf(m, s);
            float corr = __expf(m - mn);
            float e = __expf(s - mn);
            #pragma unroll
            for (int d = 0; d < DH_; ++d) acc[d] = acc[d]*corr + e*vb[j*DH_ + d];
            lsum = lsum*corr + e; m = mn;
        }
        float rin = 1.f / lsum;
        #pragma unroll
        for (int d = 0; d < DH_; ++d)
            z[((size_t)(b*L_ + i)*H_ + h)*DH_ + d] = __float2bfloat16(acc[d]*rin);
    }
}
__global__ void out_k(const float* __restrict__ t, float* __restrict__ out) {
    size_t tot = (size_t)B_*P_*D_;
    for (size_t i = (size_t)blockIdx.x*blockDim.x + threadIdx.x; i < tot; i += (size_t)gridDim.x*blockDim.x) {
        int d = (int)(i % D_); size_t r = i / D_;
        int p = (int)(r % P_); size_t b = r / P_;
        out[i] = t[(b*(size_t)L_ + p + 1)*(size_t)D_ + d];
    }
}

// ---------------------------------------------------------------------------
extern "C" void kernel_launch(void* const* d_in, const int* in_sizes, int n_in,
                              void* d_out, int out_size, void* d_ws, size_t ws_size,
                              hipStream_t stream)
{
    (void)in_sizes; (void)n_in; (void)out_size; (void)ws_size;
    const float* x    = (const float*)d_in[0];
    const float* pos  = (const float*)d_in[1];
    const float* w1   = (const float*)d_in[2];  const float* g1 = (const float*)d_in[3];  const float* b1 = (const float*)d_in[4];
    const float* w2   = (const float*)d_in[5];  const float* g2 = (const float*)d_in[6];  const float* b2 = (const float*)d_in[7];
    const float* w3   = (const float*)d_in[8];  const float* g3 = (const float*)d_in[9];  const float* b3 = (const float*)d_in[10];
    const float* w4   = (const float*)d_in[11]; const float* g4 = (const float*)d_in[12]; const float* b4 = (const float*)d_in[13];
    const float* w5   = (const float*)d_in[14]; const float* g5 = (const float*)d_in[15]; const float* b5 = (const float*)d_in[16];
    const float* cls  = (const float*)d_in[17];
    const float* ln1g = (const float*)d_in[18]; const float* ln1b = (const float*)d_in[19];
    const float* wqkv = (const float*)d_in[20];
    const float* wout = (const float*)d_in[21]; const float* bout = (const float*)d_in[22];
    const float* ln2g = (const float*)d_in[23]; const float* ln2b = (const float*)d_in[24];
    const float* wff1 = (const float*)d_in[25]; const float* bff1 = (const float*)d_in[26];
    const float* wff2 = (const float*)d_in[27]; const float* bff2 = (const float*)d_in[28];

    char* wp = (char*)d_ws;
    auto alloc = [&](size_t bytes) -> void* {
        void* p = (void*)wp; wp += (bytes + 255) & ~(size_t)255; return p;
    };
    bf16*  xb32  = (bf16*) alloc((size_t)B_*N_*32*sizeof(bf16));   // K padded 3 -> 32
    float* gram  = (float*)alloc((size_t)B_*N_*N_*sizeof(float));
    float* diag  = (float*)alloc((size_t)B_*N_*sizeof(float));
    int*   idx   = (int*)  alloc((size_t)B_*N_*KN_*sizeof(int));
    bf16*  F     = (bf16*) alloc((size_t)B_*N_*KN_*256*sizeof(bf16));
    float* Y     = (float*)alloc((size_t)B_*N_*KN_*256*sizeof(float));
    float* x1f   = (float*)alloc((size_t)B_*N_*64*sizeof(float));
    float* x2f   = (float*)alloc((size_t)B_*N_*64*sizeof(float));
    float* x3f   = (float*)alloc((size_t)B_*N_*128*sizeof(float));
    bf16*  hcat  = (bf16*) alloc((size_t)B_*N_*512*sizeof(bf16));
    float* y5    = (float*)alloc((size_t)B_*N_*D_*sizeof(float));
    float* sums  = (float*)alloc((size_t)2*D_*sizeof(float));
    float* scale = (float*)alloc((size_t)D_*sizeof(float));
    float* shift = (float*)alloc((size_t)D_*sizeof(float));
    float* t     = (float*)alloc((size_t)LP_*D_*sizeof(float));     // 528 rows (8 junk)
    bf16*  hn    = (bf16*) alloc((size_t)LP_*D_*sizeof(bf16));
    float* qkvb  = (float*)alloc((size_t)LP_*3*INNER_*sizeof(float));
    bf16*  z     = (bf16*) alloc((size_t)LP_*INNER_*sizeof(bf16));
    float* ff    = (float*)alloc((size_t)LP_*MLP_*sizeof(float));
    bf16*  ffb   = (bf16*) alloc((size_t)LP_*MLP_*sizeof(bf16));
    bf16*  w1b   = (bf16*) alloc((size_t)64*32*sizeof(bf16));       // K padded 6 -> 32
    bf16*  w2b   = (bf16*) alloc((size_t)64*128*sizeof(bf16));
    bf16*  w3b   = (bf16*) alloc((size_t)128*128*sizeof(bf16));
    bf16*  w4b   = (bf16*) alloc((size_t)256*256*sizeof(bf16));
    bf16*  w5b   = (bf16*) alloc((size_t)D_*512*sizeof(bf16));
    bf16*  wqkvT = (bf16*) alloc((size_t)DEPTH_*3*INNER_*D_*sizeof(bf16));
    bf16*  woutT = (bf16*) alloc((size_t)DEPTH_*D_*INNER_*sizeof(bf16));
    bf16*  wff1T = (bf16*) alloc((size_t)DEPTH_*MLP_*D_*sizeof(bf16));
    bf16*  wff2T = (bf16*) alloc((size_t)DEPTH_*D_*MLP_*sizeof(bf16));

    auto cdiv = [](size_t n) { return (unsigned)((n + 255) / 256); };
    // M%16==0, N%64==0, K%32==0 guaranteed by padding
    auto gemm = [&](const bf16* A, int lda, long sA, const bf16* BT, int ldb, long sB,
                    float* C, int ldc, long sC, const float* Cin, const float* bias,
                    bf16* Dbf, int M, int Nn, int K, int act, int batch) {
        dim3 grid((unsigned)((M/16 + 7)/8), (unsigned)(Nn/64), (unsigned)batch);
        gemm_k<<<grid, 256, 0, stream>>>(A, lda, sA, BT, ldb, sB, C, ldc, sC,
                                         Cin, bias, Dbf, M, Nn, K, act);
    };

    // ---- weight/input prep (cast + pad + transpose to [N,K] bf16) ----
    castpad_k<<<cdiv((size_t)B_*N_*32), 256, 0, stream>>>(x, xb32, B_*N_, 3, 32);
    castpad_k<<<cdiv((size_t)64*32),    256, 0, stream>>>(w1, w1b, 64, 6, 32);
    cast_k<<<cdiv(64*128),  256, 0, stream>>>(w2, w2b, 64*128);
    cast_k<<<cdiv(128*128), 256, 0, stream>>>(w3, w3b, 128*128);
    cast_k<<<cdiv(256*256), 256, 0, stream>>>(w4, w4b, 256*256);
    cast_k<<<cdiv((size_t)D_*512), 256, 0, stream>>>(w5, w5b, (size_t)D_*512);
    for (int l = 0; l < DEPTH_; ++l) {
        trcast_k<<<cdiv((size_t)D_*3*INNER_), 256, 0, stream>>>(wqkv + (size_t)l*D_*3*INNER_, wqkvT + (size_t)l*3*INNER_*D_, D_, 3*INNER_);
        trcast_k<<<cdiv((size_t)INNER_*D_),   256, 0, stream>>>(wout + (size_t)l*INNER_*D_,   woutT + (size_t)l*D_*INNER_,   INNER_, D_);
        trcast_k<<<cdiv((size_t)D_*MLP_),     256, 0, stream>>>(wff1 + (size_t)l*D_*MLP_,     wff1T + (size_t)l*MLP_*D_,     D_, MLP_);
        trcast_k<<<cdiv((size_t)MLP_*D_),     256, 0, stream>>>(wff2 + (size_t)l*MLP_*D_,     wff2T + (size_t)l*D_*MLP_,     MLP_, D_);
    }

    // ---- EdgeConv: knn(gram GEMM) -> gather -> GEMM -> BN -> leaky -> max ----
    // fld = padded feature width (lda of F and of the weight)
    auto edgeconv = [&](const bf16* xb, int ldx, const float* xf_in, int C, int Kpad, int fld,
                        const bf16* wb, const float* gg, const float* bb,
                        int O, float* xf_out, bf16* hslice) {
        gemm(xb, ldx, (long)N_*ldx, xb, ldx, (long)N_*ldx,
             gram, N_, (long)N_*N_, nullptr, nullptr, nullptr, N_, N_, Kpad, 0, B_);
        diag_k<<<cdiv((size_t)B_*N_), 256, 0, stream>>>(gram, diag);
        topk_k<<<cdiv((size_t)B_*N_), 256, 0, stream>>>(gram, diag, idx);
        if (fld != 2*C)  // zero pad region once per call before scattering features
            zero_k<<<cdiv((size_t)B_*N_*KN_*fld/2), 256, 0, stream>>>((float*)F, (size_t)B_*N_*KN_*fld/2);
        graph_k<<<cdiv((size_t)B_*N_*KN_*C), 256, 0, stream>>>(xf_in, idx, F, C, fld);
        gemm(F, fld, 0, wb, fld, 0, Y, O, 0, nullptr, nullptr, nullptr,
             B_*N_*KN_, O, fld, 0, 1);
        zero_k<<<cdiv((size_t)2*O), 256, 0, stream>>>(sums, (size_t)2*O);
        bnpart_k<<<dim3((unsigned)O, 32), 256, 0, stream>>>(Y, (long)B_*N_*KN_, O, sums);
        bnfin_k<<<(O + 255)/256, 256, 0, stream>>>(sums, gg, bb, scale, shift, O, (float)((long)B_*N_*KN_));
        efuse_k<<<cdiv((size_t)B_*N_*O), 256, 0, stream>>>(Y, scale, shift, xf_out, hslice, O);
    };

    edgeconv(xb32,       32,  x,   3,   32,  32,  w1b, g1, b1, 64,  x1f,     hcat + 0);
    edgeconv(hcat + 0,   512, x1f, 64,  64,  128, w2b, g2, b2, 64,  x2f,     hcat + 64);
    edgeconv(hcat + 64,  512, x2f, 64,  64,  128, w3b, g3, b3, 128, x3f,     hcat + 128);
    edgeconv(hcat + 128, 512, x3f, 128, 128, 256, w4b, g4, b4, 256, nullptr, hcat + 256);

    // ---- conv5 + BN + leaky + patch max-pool + cls ----
    gemm(hcat, 512, 0, w5b, 512, 0, y5, D_, 0, nullptr, nullptr, nullptr,
         B_*N_, D_, 512, 0, 1);
    zero_k<<<cdiv((size_t)2*D_), 256, 0, stream>>>(sums, (size_t)2*D_);
    bnpart_k<<<dim3((unsigned)D_, 32), 256, 0, stream>>>(y5, (long)B_*N_, D_, sums);
    bnfin_k<<<(D_ + 255)/256, 256, 0, stream>>>(sums, g5, b5, scale, shift, D_, (float)((long)B_*N_));
    pool_k<<<cdiv((size_t)B_*P_*D_), 256, 0, stream>>>(y5, scale, shift, t);
    cls_k<<<cdiv((size_t)B_*D_), 256, 0, stream>>>(cls, t);

    // ---- transformer (M padded 520 -> 528; junk rows never read back) ----
    for (int l = 0; l < DEPTH_; ++l) {
        addpos_k<<<cdiv((size_t)B_*L_*D_), 256, 0, stream>>>(t, pos);
        ln_k<<<B_*L_, 256, 0, stream>>>(t, ln1g + (size_t)l*D_, ln1b + (size_t)l*D_, hn);
        gemm(hn, D_, 0, wqkvT + (size_t)l*3*INNER_*D_, D_, 0, qkvb, 3*INNER_, 0,
             nullptr, nullptr, nullptr, LP_, 3*INNER_, D_, 0, 1);
        attn_k<<<B_*H_, 128, 0, stream>>>(qkvb, z);
        gemm(z, INNER_, 0, woutT + (size_t)l*D_*INNER_, INNER_, 0, t, D_, 0,
             t, bout + (size_t)l*D_, nullptr, LP_, D_, INNER_, 0, 1);
        ln_k<<<B_*L_, 256, 0, stream>>>(t, ln2g + (size_t)l*D_, ln2b + (size_t)l*D_, hn);
        gemm(hn, D_, 0, wff1T + (size_t)l*MLP_*D_, D_, 0, ff, MLP_, 0,
             nullptr, bff1 + (size_t)l*MLP_, ffb, LP_, MLP_, D_, 2, 1);
        gemm(ffb, MLP_, 0, wff2T + (size_t)l*D_*MLP_, MLP_, 0, t, D_, 0,
             t, bff2 + (size_t)l*D_, nullptr, LP_, D_, MLP_, 0, 1);
    }
    out_k<<<cdiv((size_t)B_*P_*D_), 256, 0, stream>>>(t, (float*)d_out);
}
